// MSDNet_29394756174543
// MI455X (gfx1250) — matrix operand order repaired and checked by reading, not compiled, not run on hardware
//
#include <hip/hip_runtime.h>

// ---------------------------------------------------------------------------
// MSDNet early-exit net, scale-2 chain only (scales 0/1 are dead code).
// f16 WMMA implicit-GEMM conv3x3: full-layer weights resident in LDS (288KB of
// the MI455X's 320KB/WGP), 64-position output tile per block, async global->LDS
// staging (ASYNCcnt path), fused ReLU + global-average-pool.
// ---------------------------------------------------------------------------

typedef __attribute__((ext_vector_type(16))) _Float16 v16h;
typedef __attribute__((ext_vector_type(8)))  _Float16 v8h;
typedef __attribute__((ext_vector_type(8)))  float    v8f;
typedef int v4i __attribute__((vector_size(16)));   // matches builtin param type

#define AS1 __attribute__((address_space(1)))
#define AS3 __attribute__((address_space(3)))

#define Bb   16
#define Hh   112
#define Ww   112
#define Cc   128
#define NCc  100
#define KTOT 1152                      // 9 taps * 128 cin

#define TROWS 4                        // output rows per block
#define LDS_W_BYTES (Cc * KTOT * 2)            // 294912 B packed f16 weights
#define LDS_I_BYTES ((TROWS + 2) * 18 * Cc * 2) // 27648 B haloed input tile
#define LDS_P_OFF   (LDS_W_BYTES + LDS_I_BYTES)
#define LDS_TOTAL   (LDS_P_OFF + 512)  // + 128-float pool-reduction scratch

#if __has_builtin(__builtin_amdgcn_global_load_async_to_lds_b128)
#define MSD_ASYNC 1
#else
#define MSD_ASYNC 0
#endif

__device__ __forceinline__ void msd_wait_async0() {
#if __has_builtin(__builtin_amdgcn_s_wait_asynccnt)
  __builtin_amdgcn_s_wait_asynccnt(0);
#else
  asm volatile("s_wait_asynccnt 0x0" ::: "memory");
#endif
}

// --------------------------- init conv (3 -> 128) ---------------------------
__global__ __launch_bounds__(256) void msd_init_conv(
    const float* __restrict__ xin,   // [B,3,112,112] NCHW f32
    const float* __restrict__ iw,    // [S,C,3,3,3] f32, use s=2
    const float* __restrict__ ib,    // [S,C] f32
    _Float16* __restrict__ fout)     // [B,112,112,128] NHWC f16
{
  size_t idx = (size_t)blockIdx.x * 256 + threadIdx.x;  // < 25,690,112
  int c  = (int)(idx & 127);
  size_t pos = idx >> 7;
  int xg = (int)(pos % Ww);
  size_t t2 = pos / Ww;
  int yg = (int)(t2 % Hh);
  int b  = (int)(t2 / Hh);

  float s = ib[2 * Cc + c];
  #pragma unroll
  for (int cin = 0; cin < 3; ++cin) {
    #pragma unroll
    for (int ky = 0; ky < 3; ++ky) {
      int gy = yg + ky - 1;
      if (gy < 0 || gy >= Hh) continue;
      #pragma unroll
      for (int kx = 0; kx < 3; ++kx) {
        int gx = xg + kx - 1;
        if (gx < 0 || gx >= Ww) continue;
        s += xin[(((size_t)b * 3 + cin) * Hh + gy) * Ww + gx] *
             iw[(((size_t)(2 * Cc) + c) * 3 + cin) * 9 + ky * 3 + kx];
      }
    }
  }
  fout[idx] = (_Float16)s;   // no activation on init block (matches reference)
}

// ------------------- repack layer weights to [cout][tap*128+cin] f16 --------
__global__ __launch_bounds__(256) void msd_pack_weights(
    const float* __restrict__ bw,    // [D,S,C,C,3,3] f32
    _Float16* __restrict__ wpack,    // [128][1152] f16
    int layer)
{
  int idx = blockIdx.x * 256 + threadIdx.x;   // < 147456
  int cout = idx / KTOT;
  int k    = idx % KTOT;
  int tap  = k >> 7;
  int cin  = k & 127;
  size_t src = ((((size_t)(layer * 3 + 2) * Cc + cout) * Cc + cin) * 9 + tap);
  wpack[idx] = (_Float16)bw[src];
}

// -------- conv3x3 128->128 + ReLU + fused pooling, WMMA implicit GEMM -------
__global__ __launch_bounds__(256) void msd_conv_block(
    const _Float16* __restrict__ fin,   // NHWC f16
    _Float16* __restrict__ fout,        // NHWC f16
    const _Float16* __restrict__ wpack, // [128][1152] f16
    const float* __restrict__ bias,     // [128] f32 (block_b[i,2])
    float* __restrict__ pooled)         // [B,128] f32 accumulator (pre-zeroed)
{
  extern __shared__ char smem[];
  _Float16* wl = (_Float16*)smem;                  // 128 x 1152 weights
  _Float16* il = (_Float16*)(smem + LDS_W_BYTES);  // 6 x 18 x 128 input tile
  float*    pl = (float*)(smem + LDS_P_OFF);       // 128 pool partials

  const int bid = blockIdx.x;        // B * 28 * 7 = 3136 blocks
  const int b   = bid / 196;
  const int tt  = bid % 196;
  const int y0  = (tt / 7) * TROWS;  // output row quad
  const int x0  = (tt % 7) * 16;     // output col tile
  const int tid = threadIdx.x;

  // --- stage full layer weights (288 KB, L2-resident) + haloed input tile ---
#if MSD_ASYNC
  {
    const char* gsrc = (const char*)wpack;
    AS3 char* ldst = (AS3 char*)smem;
    #pragma unroll 4
    for (int i = tid; i < LDS_W_BYTES / 16; i += 256)
      __builtin_amdgcn_global_load_async_to_lds_b128(
          (AS1 v4i*)(gsrc + (size_t)i * 16),
          (AS3 v4i*)(ldst + (size_t)i * 16), 0, 0);
    for (int i = tid; i < LDS_I_BYTES / 16; i += 256) {
      int pos = i >> 4;              // (row 0..5, col 0..17)
      int cc  = i & 15;              // 16B chunk of the 128 channels
      int r = pos / 18, c = pos % 18;
      int gy = y0 - 1 + r, gx = x0 - 1 + c;
      if (gy >= 0 && gy < Hh && gx >= 0 && gx < Ww) {
        const char* g = (const char*)(fin + (size_t)((b * Hh + gy) * Ww + gx) * Cc)
                        + (size_t)cc * 16;
        __builtin_amdgcn_global_load_async_to_lds_b128(
            (AS1 v4i*)g,
            (AS3 v4i*)((AS3 char*)smem + LDS_W_BYTES + (size_t)i * 16), 0, 0);
      } else {
        ((uint4*)il)[i] = make_uint4(0u, 0u, 0u, 0u);  // padding (disjoint addrs)
      }
    }
  }
  if (tid < 128) pl[tid] = 0.f;
  msd_wait_async0();
  __syncthreads();
#else
  {
    const uint4* src = (const uint4*)wpack;
    uint4* dst = (uint4*)wl;
    #pragma unroll 4
    for (int i = tid; i < LDS_W_BYTES / 16; i += 256) dst[i] = src[i];
    uint4* dsti = (uint4*)il;
    for (int i = tid; i < LDS_I_BYTES / 16; i += 256) {
      int pos = i >> 4;
      int cc  = i & 15;
      int r = pos / 18, c = pos % 18;
      int gy = y0 - 1 + r, gx = x0 - 1 + c;
      uint4 v = make_uint4(0u, 0u, 0u, 0u);
      if (gy >= 0 && gy < Hh && gx >= 0 && gx < Ww)
        v = ((const uint4*)(fin + (size_t)((b * Hh + gy) * Ww + gx) * Cc))[cc];
      dsti[i] = v;
    }
  }
  if (tid < 128) pl[tid] = 0.f;
  __syncthreads();
#endif

  const int lane = tid & 31;
  const int wv   = tid >> 5;         // 8 waves: cout tile = wv
  const int lm   = lane & 15;
  const int hi   = lane >> 4;        // 0 = lanes 0-15, 1 = lanes 16-31

  v8f acc[TROWS];
  #pragma unroll
  for (int j = 0; j < TROWS; ++j) acc[j] = v8f{};

  // A frag, lane view: M = lm, halves 0-15 cover
  // K = kk + 16*hi .. +15 (contiguous, matches B).
  const int wrow = (wv * 16 + lm) * KTOT + hi * 16;

  #pragma unroll   // full 36-step K-loop: immediate ds offsets + deep pipelining
  for (int tap = 0; tap < 9; ++tap) {
    const int dy = tap / 3, dx = tap % 3;
    const int ibase = (lm + dx) * Cc + hi * 16;
    #pragma unroll
    for (int ch = 0; ch < 4; ++ch) {
      const int kk = tap * Cc + ch * 32;
      v8h alo = *(const v8h*)(wl + wrow + kk);
      v8h ahi = *(const v8h*)(wl + wrow + kk + 8);
      v16h a = __builtin_shufflevector(alo, ahi,
                 0,1,2,3,4,5,6,7,8,9,10,11,12,13,14,15);
      // B frag (contiguous k order, same as A): lane = column N, lanes 16-31 take
      // K+16 -> one contiguous 32B LDS read per lane. A reused across 4 rows.
      #pragma unroll
      for (int j = 0; j < TROWS; ++j) {
        v16h bf = *(const v16h*)(il + ((dy + j) * 18) * Cc + ibase + ch * 32);
        acc[j] = __builtin_amdgcn_wmma_f32_16x16x32_f16(
                     false, a, false, bf, (short)0, acc[j], false, false);
      }
    }
  }

  // --- epilogue: bias + ReLU, NHWC b128 store, fused pooling reduction ---
  // C/D layout: VGPR r, lane l -> M = r + 8*(l>=16), N = l%16.
  const int coutBase = wv * 16 + hi * 8;
  float bs[8];
  #pragma unroll
  for (int r = 0; r < 8; ++r) bs[r] = bias[coutBase + r];

  #pragma unroll
  for (int j = 0; j < TROWS; ++j) {
    v8h st;
    #pragma unroll
    for (int r = 0; r < 8; ++r) {
      float v = acc[j][r] + bs[r];
      v = v > 0.f ? v : 0.f;
      st[r] = (_Float16)v;
      atomicAdd(&pl[coutBase + r], v);          // ds_add_f32
    }
    *(v8h*)(fout + (size_t)((b * Hh + (y0 + j)) * Ww + (x0 + lm)) * Cc
            + coutBase) = st;
  }
  __syncthreads();
  if (tid < 128) atomicAdd(&pooled[b * Cc + tid], pl[tid]);  // global f32 atomic
}

// ---------------- classifier + softmax confidence + early exit --------------
__global__ __launch_bounds__(128) void msd_classify(
    const float* __restrict__ pooled,  // [B,128] channel sums
    const float* __restrict__ clw,     // [D,100,128]
    const float* __restrict__ clb,     // [D,100]
    float* __restrict__ dout,          // [B,100]
    int* __restrict__ exited,          // [B]
    float* __restrict__ poolClear,     // == pooled; zeroed for next layer
    int layer)
{
  __shared__ float lg[Bb * NCc];
  const int t = threadIdx.x;

  for (int id = t; id < Bb * NCc; id += 128) {
    int b = id / NCc, n = id % NCc;
    const float* wr = clw + ((size_t)layer * NCc + n) * Cc;
    const float* pr = pooled + b * Cc;
    float s = 0.f;
    #pragma unroll 4
    for (int c = 0; c < Cc; ++c) s += pr[c] * wr[c];
    lg[id] = s * (1.0f / (Hh * Ww)) + clb[layer * NCc + n];
  }
  __syncthreads();

  if (t < Bb) {
    int b = t;
    if (!exited[b]) {
      float m = -3.4e38f;
      for (int n = 0; n < NCc; ++n) m = fmaxf(m, lg[b * NCc + n]);
      float se = 0.f;
      for (int n = 0; n < NCc; ++n) se += expf(lg[b * NCc + n] - m);
      float conf = 1.f / se;               // max softmax prob
      if (conf >= 0.9f || layer == 3) {    // layer-3 logits == final_logits
        for (int n = 0; n < NCc; ++n) dout[b * NCc + n] = lg[b * NCc + n];
        exited[b] = 1;
      }
    }
  }
  __syncthreads();
  for (int id = t; id < Bb * Cc; id += 128) poolClear[id] = 0.f;
}

// ------------------------------- state zeroing ------------------------------
__global__ void msd_zero(float* pooled, int* exited) {
  int t = blockIdx.x * 256 + threadIdx.x;
  if (t < Bb * Cc) pooled[t] = 0.f;
  if (t < Bb)      exited[t] = 0;
}

// ---------------------------------------------------------------------------
extern "C" void kernel_launch(void* const* d_in, const int* in_sizes, int n_in,
                              void* d_out, int out_size, void* d_ws, size_t ws_size,
                              hipStream_t stream) {
  const float* x       = (const float*)d_in[0];
  const float* init_w  = (const float*)d_in[1];
  const float* init_b  = (const float*)d_in[2];
  const float* block_w = (const float*)d_in[3];
  const float* block_b = (const float*)d_in[4];
  const float* cls_w   = (const float*)d_in[5];
  const float* cls_b   = (const float*)d_in[6];
  float* out = (float*)d_out;

  const size_t featElems = (size_t)Bb * Hh * Ww * Cc;   // 25,690,112
  _Float16* featA = (_Float16*)d_ws;
  _Float16* featB = featA + featElems;
  _Float16* wpack = featB + featElems;
  float* pooled   = (float*)(wpack + (size_t)Cc * KTOT);
  int*   exited   = (int*)(pooled + Bb * Cc);

  (void)hipFuncSetAttribute((const void*)msd_conv_block,
                            hipFuncAttributeMaxDynamicSharedMemorySize,
                            LDS_TOTAL);

  msd_zero<<<8, 256, 0, stream>>>(pooled, exited);
  msd_init_conv<<<(unsigned)(featElems / 256), 256, 0, stream>>>(
      x, init_w, init_b, featA);

  for (int i = 0; i < 4; ++i) {
    msd_pack_weights<<<576, 256, 0, stream>>>(block_w, wpack, i);
    const _Float16* fi = (i & 1) ? featB : featA;
    _Float16*       fo = (i & 1) ? featA : featB;
    msd_conv_block<<<Bb * 196, 256, LDS_TOTAL, stream>>>(
        fi, fo, wpack, block_b + (size_t)(i * 3 + 2) * Cc, pooled);
    msd_classify<<<1, 128, 0, stream>>>(pooled, cls_w, cls_b, out, exited,
                                        pooled, i);
  }
}
